// GATLayer_4638564679689
// MI455X (gfx1250) — compile-verified, matrix operand order checked by reading, not run on hardware
//
#include <hip/hip_runtime.h>
#include <hip/hip_bf16.h>
#include <stdint.h>

// GATConv (PyG-style) for MI455X / gfx1250.
//  - he = ea@W_e folded into a_edge = ea @ (W_e . att_edge)  ([64,4] matrix) -> kills 28 GFLOP + ~1.7GB traffic
//  - self-loop attr term = segment_sum(a_edge)/deg (linearity) -> no [N,64] loop_attr tensor
//  - h = x@W done with v_wmma_f32_16x16x32_bf16 (f32 accumulate); full-tile fast path (no per-store exec churn)
//  - segment softmax: ordered-bits atomicMax (self loop initializes every node) + atomicAdd
//  - aggregation: warp-per-edge gather of h[src] (1KB) + f32 atomic scatter (out is L2-resident: 51MB << 192MB L2)

#define HEADS 4
#define CCH   64
#define INCH  64
#define HC    256
#define SLOPE 0.2f
#define EPSV  1e-16f

typedef __attribute__((ext_vector_type(16))) __bf16    v16bf;
typedef __attribute__((ext_vector_type(8)))  float     v8f;
typedef __attribute__((ext_vector_type(4)))  unsigned  u32x4;

union FragU { v16bf v; u32x4 q[2]; };

__device__ __forceinline__ float lrelu(float x) { return x > 0.f ? x : SLOPE * x; }
// order-preserving float<->uint mapping for atomicMax on floats
__device__ __forceinline__ unsigned ordF(float f) {
    unsigned u = __float_as_uint(f);
    return (u & 0x80000000u) ? ~u : (u | 0x80000000u);
}
__device__ __forceinline__ float unordF(unsigned u) {
    unsigned v = (u & 0x80000000u) ? (u & 0x7FFFFFFFu) : ~u;
    return __uint_as_float(v);
}

__global__ void k_zero(float* p, int n) {
    int i = blockIdx.x * blockDim.x + threadIdx.x;
    if (i < n) p[i] = 0.f;
}

// one block of 256 threads: fold W_e with att_edge -> vedge[64][4]; transpose W -> bf16 Wt[256][64]
__global__ void k_prep(const float* __restrict__ We, const float* __restrict__ attE,
                       const float* __restrict__ Wf, float* __restrict__ vedge,
                       __bf16* __restrict__ wt) {
    int t = threadIdx.x;             // 0..255
    int k = t >> 2, hh = t & 3;
    float s = 0.f;
    for (int c = 0; c < CCH; ++c) s += We[k * HC + hh * CCH + c] * attE[hh * CCH + c];
    vedge[k * HEADS + hh] = s;
    for (int kk = 0; kk < INCH; ++kk) wt[t * INCH + kk] = (__bf16)Wf[kk * HC + t];
}

__global__ void k_xconv(const float* __restrict__ x, __bf16* __restrict__ xb, int n) {
    int i = blockIdx.x * blockDim.x + threadIdx.x;
    if (i < n) xb[i] = (__bf16)x[i];
}

// h = x @ W : wave computes a 16-row tile across all 16 column tiles, K=64 as 2 WMMAs.
// Full tiles (the only case for N=50000) take an unguarded store path.
__global__ void k_gemm(const __bf16* __restrict__ xb, const __bf16* __restrict__ wt,
                       float* __restrict__ h, int N_, int Mtiles) {
    int wave = (blockIdx.x * blockDim.x + threadIdx.x) >> 5;
    if (wave >= Mtiles) return;                 // uniform per wave: EXEC stays all-ones for WMMA
    int lane = threadIdx.x & 31;
    int r = lane & 15, hs = lane >> 4;
    int m0 = wave * 16;
    bool full = (m0 + 16 <= N_);                // wave-uniform
    int row = m0 + r; if (row >= N_) row = N_ - 1;
    const __bf16* ar = xb + (size_t)row * INCH;
    FragU A0, A1;
    A0.q[0] = *reinterpret_cast<const u32x4*>(ar + 8 * hs);        // K = 8hs .. 8hs+7
    A0.q[1] = *reinterpret_cast<const u32x4*>(ar + 16 + 8 * hs);   // K = 16+8hs ..
    A1.q[0] = *reinterpret_cast<const u32x4*>(ar + 32 + 8 * hs);
    A1.q[1] = *reinterpret_cast<const u32x4*>(ar + 48 + 8 * hs);
    for (int nt = 0; nt < 16; ++nt) {
        const __bf16* br = wt + (size_t)(nt * 16 + r) * INCH;      // Wt row = column of W
        FragU B0, B1;
        B0.q[0] = *reinterpret_cast<const u32x4*>(br + 8 * hs);
        B0.q[1] = *reinterpret_cast<const u32x4*>(br + 16 + 8 * hs);
        B1.q[0] = *reinterpret_cast<const u32x4*>(br + 32 + 8 * hs);
        B1.q[1] = *reinterpret_cast<const u32x4*>(br + 48 + 8 * hs);
        v8f c = {};
        c = __builtin_amdgcn_wmma_f32_16x16x32_bf16(false, A0.v, false, B0.v, (short)0, c, false, false);
        c = __builtin_amdgcn_wmma_f32_16x16x32_bf16(false, A1.v, false, B1.v, (short)0, c, false, false);
        // D layout: M = vgpr_idx + 8*(lane>=16), N = lane&15
        float* hp = h + (size_t)(m0 + hs * 8) * HC + nt * 16 + r;
        if (full) {
#pragma unroll
            for (int i = 0; i < 8; ++i) hp[(size_t)i * HC] = c[i];
        } else {
#pragma unroll
            for (int i = 0; i < 8; ++i)
                if (m0 + hs * 8 + i < N_) hp[(size_t)i * HC] = c[i];
        }
    }
}

// per-node a_src/a_dst = h[n] . att  (warp per node; 8 lanes per head, shuffle-reduce)
__global__ void k_attn_nodes(const float* __restrict__ h, const float* __restrict__ attS,
                             const float* __restrict__ attD, float* __restrict__ asrc,
                             float* __restrict__ adst, int N_) {
    int wid = (blockIdx.x * blockDim.x + threadIdx.x) >> 5;
    if (wid >= N_) return;
    int lane = threadIdx.x & 31;
    int head = lane >> 3;
    int c0 = lane * 8;
    float ss = 0.f, sd = 0.f;
    const float* hp = h + (size_t)wid * HC + c0;
    const float* as = attS + head * CCH + (lane & 7) * 8;
    const float* ad = attD + head * CCH + (lane & 7) * 8;
#pragma unroll
    for (int i = 0; i < 8; ++i) { float v = hp[i]; ss += v * as[i]; sd += v * ad[i]; }
#pragma unroll
    for (int off = 1; off < 8; off <<= 1) { ss += __shfl_xor(ss, off); sd += __shfl_xor(sd, off); }
    if ((lane & 7) == 0) { asrc[wid * HEADS + head] = ss; adst[wid * HEADS + head] = sd; }
}

// warp per edge: a_edge[e,:] = ea[e,:] @ vedge ; accumulate deg + per-dst sum (for self-loop term)
__global__ void k_edge_ae(const float* __restrict__ ea, const long long* __restrict__ ei,
                          const float* __restrict__ vedge, float* __restrict__ elog,
                          float* __restrict__ deg, float* __restrict__ aeloop, int E_) {
    __shared__ float sv[INCH * HEADS];
    for (int i = threadIdx.x; i < INCH * HEADS; i += blockDim.x) sv[i] = vedge[i];
    __syncthreads();
    int wid = (blockIdx.x * blockDim.x + threadIdx.x) >> 5;
    if (wid >= E_) return;
    int lane = threadIdx.x & 31;
    float p0 = 0.f, p1 = 0.f, p2 = 0.f, p3 = 0.f;
    const float* er = ea + (size_t)wid * INCH;
#pragma unroll
    for (int kk = 0; kk < 2; ++kk) {
        int k = lane + kk * 32;
        float v = er[k];
        p0 += v * sv[k * HEADS + 0]; p1 += v * sv[k * HEADS + 1];
        p2 += v * sv[k * HEADS + 2]; p3 += v * sv[k * HEADS + 3];
    }
#pragma unroll
    for (int off = 1; off < 32; off <<= 1) {
        p0 += __shfl_xor(p0, off); p1 += __shfl_xor(p1, off);
        p2 += __shfl_xor(p2, off); p3 += __shfl_xor(p3, off);
    }
    if (lane == 0) {
        int dst = (int)ei[E_ + wid];
        elog[wid * HEADS + 0] = p0; elog[wid * HEADS + 1] = p1;
        elog[wid * HEADS + 2] = p2; elog[wid * HEADS + 3] = p3;
        atomicAdd(deg + dst, 1.0f);
        atomicAdd(aeloop + dst * HEADS + 0, p0);
        atomicAdd(aeloop + dst * HEADS + 1, p1);
        atomicAdd(aeloop + dst * HEADS + 2, p2);
        atomicAdd(aeloop + dst * HEADS + 3, p3);
    }
}

// self-loop logit per node; initializes segment-max (every node has a self loop)
__global__ void k_node_loop(const float* __restrict__ deg, const float* __restrict__ aeloop,
                            const float* __restrict__ asrc, const float* __restrict__ adst,
                            float* __restrict__ llog, unsigned* __restrict__ mb, int N_) {
    int n = blockIdx.x * blockDim.x + threadIdx.x;
    if (n >= N_) return;
    float d = fmaxf(deg[n], 1.0f);
#pragma unroll
    for (int hh = 0; hh < HEADS; ++hh) {
        float ll = lrelu(asrc[n * HEADS + hh] + adst[n * HEADS + hh] + aeloop[n * HEADS + hh] / d);
        llog[n * HEADS + hh] = ll;
        mb[n * HEADS + hh] = ordF(ll);
    }
}

__global__ void k_edge_max(const long long* __restrict__ ei, const float* __restrict__ asrc,
                           const float* __restrict__ adst, float* __restrict__ elog,
                           unsigned* __restrict__ mb, int E_) {
    int e = blockIdx.x * blockDim.x + threadIdx.x;
    if (e >= E_) return;
    int s = (int)ei[e], d = (int)ei[E_ + e];
#pragma unroll
    for (int hh = 0; hh < HEADS; ++hh) {
        float lg = lrelu(asrc[s * HEADS + hh] + adst[d * HEADS + hh] + elog[e * HEADS + hh]);
        elog[e * HEADS + hh] = lg;
        atomicMax(mb + d * HEADS + hh, ordF(lg));
    }
}

// finalize max to float (in place) and seed z with the self-loop exp term
__global__ void k_node_mz(unsigned* __restrict__ mb, const float* __restrict__ llog,
                          float* __restrict__ z, int N_) {
    int n = blockIdx.x * blockDim.x + threadIdx.x;
    if (n >= N_) return;
#pragma unroll
    for (int hh = 0; hh < HEADS; ++hh) {
        float m = unordF(mb[n * HEADS + hh]);
        reinterpret_cast<float*>(mb)[n * HEADS + hh] = m;
        z[n * HEADS + hh] = expf(llog[n * HEADS + hh] - m);
    }
}

__global__ void k_edge_z(const long long* __restrict__ ei, const float* __restrict__ mf,
                         float* __restrict__ elog, float* __restrict__ z, int E_) {
    int e = blockIdx.x * blockDim.x + threadIdx.x;
    if (e >= E_) return;
    int d = (int)ei[E_ + e];
#pragma unroll
    for (int hh = 0; hh < HEADS; ++hh) {
        float p = expf(elog[e * HEADS + hh] - mf[d * HEADS + hh]);
        elog[e * HEADS + hh] = p;
        atomicAdd(z + d * HEADS + hh, p);
    }
}

// out[n] = bias + alpha_selfloop * h[n]   (writes every output element -> no pre-zero needed)
__global__ void k_self_out(const float* __restrict__ h, const float* __restrict__ llog,
                           const float* __restrict__ mf, const float* __restrict__ z,
                           const float* __restrict__ bias, float* __restrict__ out, int N_) {
    int wid = (blockIdx.x * blockDim.x + threadIdx.x) >> 5;
    if (wid >= N_) return;
    int lane = threadIdx.x & 31;
    int head = lane >> 3;
    int c0 = lane * 8;
    float al = expf(llog[wid * HEADS + head] - mf[wid * HEADS + head]) / (z[wid * HEADS + head] + EPSV);
    const float* hp = h + (size_t)wid * HC + c0;
    float* op = out + (size_t)wid * HC + c0;
#pragma unroll
    for (int i = 0; i < 8; ++i) op[i] = bias[c0 + i] + al * hp[i];
}

// warp per edge: out[dst] += alpha * h[src]  (1KB gather + 256 f32 atomics; out is L2-resident)
__global__ void k_edge_agg(const long long* __restrict__ ei, const float* __restrict__ h,
                           const float* __restrict__ elog, const float* __restrict__ z,
                           float* __restrict__ out, int E_) {
    int wid = (blockIdx.x * blockDim.x + threadIdx.x) >> 5;
    if (wid >= E_) return;
    int lane = threadIdx.x & 31;
    int head = lane >> 3;
    int c0 = lane * 8;
    int s = (int)ei[wid], d = (int)ei[E_ + wid];
    float alpha = elog[wid * HEADS + head] / (z[d * HEADS + head] + EPSV);
    const float4* hp = reinterpret_cast<const float4*>(h + (size_t)s * HC + c0);
    float4 v0 = hp[0], v1 = hp[1];
    float* op = out + (size_t)d * HC + c0;
    atomicAdd(op + 0, alpha * v0.x); atomicAdd(op + 1, alpha * v0.y);
    atomicAdd(op + 2, alpha * v0.z); atomicAdd(op + 3, alpha * v0.w);
    atomicAdd(op + 4, alpha * v1.x); atomicAdd(op + 5, alpha * v1.y);
    atomicAdd(op + 6, alpha * v1.z); atomicAdd(op + 7, alpha * v1.w);
}

extern "C" void kernel_launch(void* const* d_in, const int* in_sizes, int n_in,
                              void* d_out, int out_size, void* d_ws, size_t ws_size,
                              hipStream_t stream) {
    const float*     x    = (const float*)d_in[0];
    const long long* ei   = (const long long*)d_in[1];   // int64 edge_index [2,E]
    const float*     ea   = (const float*)d_in[2];
    const float*     W    = (const float*)d_in[3];
    const float*     We   = (const float*)d_in[4];
    const float*     attS = (const float*)d_in[5];
    const float*     attD = (const float*)d_in[6];
    const float*     attE = (const float*)d_in[7];
    const float*     bias = (const float*)d_in[8];
    float*           out  = (float*)d_out;

    const int N_ = in_sizes[0] / INCH;
    const int E_ = in_sizes[1] / 2;

    // workspace carve (aligned 256B): ~75 MB total
    char* p = (char*)d_ws;
    auto carve = [&](size_t bytes) { void* r = (void*)p; p += (bytes + 255) & ~(size_t)255; return r; };
    float*    h      = (float*)carve((size_t)N_ * HC * 4);
    __bf16*   xb     = (__bf16*)carve((size_t)N_ * INCH * 2);
    __bf16*   wt     = (__bf16*)carve((size_t)HC * INCH * 2);
    float*    vedge  = (float*)carve((size_t)INCH * HEADS * 4);
    float*    asrc   = (float*)carve((size_t)N_ * HEADS * 4);
    float*    adst   = (float*)carve((size_t)N_ * HEADS * 4);
    float*    elog   = (float*)carve((size_t)E_ * HEADS * 4);
    float*    deg    = (float*)carve((size_t)N_ * 4);
    float*    aeloop = (float*)carve((size_t)N_ * HEADS * 4);
    unsigned* mb     = (unsigned*)carve((size_t)N_ * HEADS * 4);
    float*    z      = (float*)carve((size_t)N_ * HEADS * 4);
    float*    llog   = (float*)carve((size_t)N_ * HEADS * 4);

    const int TB = 256;
    // accumulators that need zeroing (deg, aeloop); z is seeded by direct store, out fully written by k_self_out
    k_zero<<<(N_ + TB - 1) / TB, TB, 0, stream>>>(deg, N_);
    k_zero<<<(N_ * HEADS + TB - 1) / TB, TB, 0, stream>>>(aeloop, N_ * HEADS);

    k_prep<<<1, 256, 0, stream>>>(We, attE, W, vedge, wt);
    k_xconv<<<((size_t)N_ * INCH + TB - 1) / TB, TB, 0, stream>>>(x, xb, N_ * INCH);

    const int Mtiles = (N_ + 15) / 16;
    k_gemm<<<(Mtiles + 7) / 8, TB, 0, stream>>>(xb, wt, h, N_, Mtiles);

    k_attn_nodes<<<(N_ + 7) / 8, TB, 0, stream>>>(h, attS, attD, asrc, adst, N_);
    k_edge_ae<<<(E_ + 7) / 8, TB, 0, stream>>>(ea, ei, vedge, elog, deg, aeloop, E_);
    k_node_loop<<<(N_ + TB - 1) / TB, TB, 0, stream>>>(deg, aeloop, asrc, adst, llog, mb, N_);
    k_edge_max<<<(E_ + TB - 1) / TB, TB, 0, stream>>>(ei, asrc, adst, elog, mb, E_);
    k_node_mz<<<(N_ + TB - 1) / TB, TB, 0, stream>>>(mb, llog, z, N_);
    k_edge_z<<<(E_ + TB - 1) / TB, TB, 0, stream>>>(ei, (const float*)mb, elog, z, E_);
    k_self_out<<<(N_ + 7) / 8, TB, 0, stream>>>(h, llog, (const float*)mb, z, bias, out, N_);
    k_edge_agg<<<(E_ + 7) / 8, TB, 0, stream>>>(ei, h, elog, z, out, E_);
}
